// MultiHeadAttention_12326556139915
// MI455X (gfx1250) — matrix operand order repaired and checked by reading, not compiled, not run on hardware
//
#include <hip/hip_runtime.h>

// ---------------------------------------------------------------------------
// Linformer attention forward for MI455X (gfx1250).
// wave32, v_wmma_f32_16x16x32_bf16, async global->LDS double buffering.
// B=4 S=4096 D=1024 H=16 DK=64 R=32
// ---------------------------------------------------------------------------

typedef __bf16 bf16_t;
typedef __attribute__((ext_vector_type(16))) __bf16 v16bf;
typedef __attribute__((ext_vector_type(8)))  float  v8f;
typedef __attribute__((ext_vector_type(4)))  int    v4i_t;
typedef __attribute__((ext_vector_type(2)))  int    v2i_t;

constexpr int CB  = 4;
constexpr int CS  = 4096;
constexpr int CD  = 1024;
constexpr int CH  = 16;
constexpr int CDK = 64;
constexpr int CR  = 32;
constexpr int CM  = CB * CS;       // 16384 rows
constexpr int CHK = CH * CDK;      // 1024
constexpr int CHR = CH * CR;       // 512

// ---- CDNA5 async global->LDS path (ASYNCcnt), with synchronous fallback ----
#if defined(__has_builtin)
#  if __has_builtin(__builtin_amdgcn_global_load_async_to_lds_b128) && \
      __has_builtin(__builtin_amdgcn_global_load_async_to_lds_b64)  && \
      __has_builtin(__builtin_amdgcn_s_wait_asynccnt)
#    define USE_ASYNC 1
#  endif
#endif
#ifndef USE_ASYNC
#  define USE_ASYNC 0
#endif

#if USE_ASYNC
#  define WAIT_ASYNC(n) __builtin_amdgcn_s_wait_asynccnt(n)
typedef __attribute__((address_space(1))) v4i_t* gp4_t;
typedef __attribute__((address_space(3))) v4i_t* lp4_t;
typedef __attribute__((address_space(1))) v2i_t* gp2_t;
typedef __attribute__((address_space(3))) v2i_t* lp2_t;
#else
#  define WAIT_ASYNC(n) ((void)0)
#endif

__device__ __forceinline__ void cp16(void* l, const void* g) {
#if USE_ASYNC
  __builtin_amdgcn_global_load_async_to_lds_b128((gp4_t)(void*)g, (lp4_t)l, 0, 0);
#else
  *(int4*)l = *(const int4*)g;
#endif
}
__device__ __forceinline__ void cp8(void* l, const void* g) {
#if USE_ASYNC
  __builtin_amdgcn_global_load_async_to_lds_b64((gp2_t)(void*)g, (lp2_t)l, 0, 0);
#else
  *(int2*)l = *(const int2*)g;
#endif
}

// --------------------------- weight conversion -----------------------------
__global__ void f32_to_bf16_kernel(const float* __restrict__ x,
                                   bf16_t* __restrict__ y, int n) {
  int i = blockIdx.x * blockDim.x + threadIdx.x;
  if (i < n) y[i] = (bf16_t)x[i];
}

// ---------------------------------------------------------------------------
// GEMM: C[M,N] = A[M,K] * W[K,N] + bias[N]
// Block tile 64x64, 8 waves, wave tile 16x32 (2 WMMA accumulators, shared A
// fragment). K-step 32, async double-buffered LDS staging.
// A stays f32 in LDS (cvt to bf16 in fragment build); W tile natural [k][n].
// A fragment (16x32): lanes 0-15 M=lane, K=0..15;
//   lanes 16-31 M=lane-16, K=16..31 (contiguous k, same as B).
// B fragment (32x16): lanes 0-15 N=lane K=0..15; lanes 16-31 N=lane-16 K=16..31
// ---------------------------------------------------------------------------
constexpr int AROW_F = 36;   // f32 elems per LDS A row (pad)
constexpr int AROW_B = 40;   // bf16 elems per LDS A row (pad)
constexpr int WROW   = 72;   // bf16 elems per LDS W row (pad)
constexpr int ABUF_BYTES = 64 * AROW_F * 4;   // 9216 B (>= 64*AROW_B*2)

template <bool A_BF16, bool OUT_BF16>
__global__ __launch_bounds__(256)
void gemm_bias_wmma(const void* __restrict__ A_,
                    const bf16_t* __restrict__ W,
                    const float* __restrict__ bias,
                    void* __restrict__ C_,
                    int M, int N, int K) {
  __shared__ __align__(16) char   AsRaw[2][ABUF_BYTES];
  __shared__ __align__(16) bf16_t Ws[2][32][WROW];

  const int tid  = threadIdx.x;
  const int lane = tid & 31;
  const int wid  = tid >> 5;
  const int hl   = lane >> 4;
  const int lr   = lane & 15;
  const int mi   = wid & 3;              // 4 M sub-tiles of 16
  const int ni   = wid >> 2;             // 2 N sub-tiles of 32
  const int m_blk = blockIdx.y * 64;
  const int n_blk = blockIdx.x * 64;

  const float*  Ag  = (const float*)A_;
  const bf16_t* Abg = (const bf16_t*)A_;

  // uniform per-thread async transfers per tile (for exact s_wait threshold)
  constexpr int TBATCH = A_BF16 ? 2 : 3;

  auto stage = [&](int kt, int p) {
    if (A_BF16) {
      // A tile 64x32 bf16 = 256 x b128
      int r = tid >> 2, qq = (tid & 3) * 8;
      bf16_t* dst = (bf16_t*)AsRaw[p];
      cp16(dst + r * AROW_B + qq, Abg + (size_t)(m_blk + r) * K + kt + qq);
    } else {
      // A tile 64x32 f32 = 512 x b128
#pragma unroll
      for (int j = 0; j < 2; ++j) {
        int c = j * 256 + tid;
        int r = c >> 3, qq = (c & 7) * 4;
        float* dst = (float*)AsRaw[p];
        cp16(dst + r * AROW_F + qq, Ag + (size_t)(m_blk + r) * K + kt + qq);
      }
    }
    // W tile 32x64 bf16 = 256 x b128, natural [k][n]
    int kk = tid >> 3, n8 = (tid & 7) * 8;
    cp16(&Ws[p][kk][n8], W + (size_t)(kt + kk) * N + n_blk + n8);
  };

  v8f acc[2] = {{}, {}};
  stage(0, 0);

  for (int kt = 0; kt < K; kt += 32) {
    const int p = (kt >> 5) & 1;
    if (kt + 32 < K) {
      stage(kt + 32, 1 - p);     // overlap next tile with this tile's WMMAs
      WAIT_ASYNC(TBATCH);        // previous batch fully landed in LDS
    } else {
      WAIT_ASYNC(0);
    }
    __syncthreads();

    // ---- A fragment (one per wave, reused for both WMMAs) ----
    v16bf a;
    const int ka = hl * 16;
    if (A_BF16) {
      const bf16_t* Al = (const bf16_t*)AsRaw[p] + (mi * 16 + lr) * AROW_B;
#pragma unroll
      for (int j = 0; j < 4; ++j) {
        a[2 * j]         = Al[ka + 2 * j];
        a[2 * j + 1]     = Al[ka + 2 * j + 1];
        a[8 + 2 * j]     = Al[ka + 8 + 2 * j];
        a[8 + 2 * j + 1] = Al[ka + 8 + 2 * j + 1];
      }
    } else {
      const float* Al = (const float*)AsRaw[p] + (mi * 16 + lr) * AROW_F;
#pragma unroll
      for (int j = 0; j < 4; ++j) {
        a[2 * j]         = (bf16_t)Al[ka + 2 * j];
        a[2 * j + 1]     = (bf16_t)Al[ka + 2 * j + 1];
        a[8 + 2 * j]     = (bf16_t)Al[ka + 8 + 2 * j];
        a[8 + 2 * j + 1] = (bf16_t)Al[ka + 8 + 2 * j + 1];
      }
    }
    // ---- two B fragments / two WMMAs ----
    const int kb = hl * 16;
#pragma unroll
    for (int sub = 0; sub < 2; ++sub) {
      const int col = ni * 32 + sub * 16 + lr;
      v16bf b;
#pragma unroll
      for (int j = 0; j < 8; ++j) {
        b[2 * j]     = Ws[p][kb + 2 * j][col];
        b[2 * j + 1] = Ws[p][kb + 2 * j + 1][col];
      }
      acc[sub] = __builtin_amdgcn_wmma_f32_16x16x32_bf16(
          false, a, false, b, (short)0, acc[sub], false, false);
    }
    __syncthreads();   // compute done before next stage overwrites 1-p
  }

  // ---- epilogue ----
#pragma unroll
  for (int sub = 0; sub < 2; ++sub) {
    const int col = n_blk + ni * 32 + sub * 16 + lr;
    const float bv = bias[col];
#pragma unroll
    for (int v = 0; v < 8; ++v) {
      int row = m_blk + mi * 16 + v + hl * 8;
      float val = acc[sub][v] + bv;
      if (OUT_BF16)
        ((bf16_t*)C_)[(size_t)row * N + col] = (bf16_t)val;
      else
        ((float*)C_)[(size_t)row * N + col] = val;
    }
  }
}

// ---------------------------------------------------------------------------
// Compression: kc[b,h,r,d] = sum_s eh[b,s,h,r]*kh[b,s,h,d]  (vc from fh,vh)
// One workgroup per (b,h); 8 waves = 2 r-tiles x 4 d-tiles; K over S in steps
// of 32 (128 iters), async double-buffered, 2 WMMAs/wave/iter.
// LDS tiles in natural [s][.] order; fragments read K-strided.
// ---------------------------------------------------------------------------
__global__ __launch_bounds__(256)
void compress_kv_wmma(const bf16_t* __restrict__ eh, const bf16_t* __restrict__ fh,
                      const bf16_t* __restrict__ kh, const bf16_t* __restrict__ vh,
                      bf16_t* __restrict__ kc, bf16_t* __restrict__ vc) {
  const int bh = blockIdx.x;
  const int b = bh / CH, h = bh % CH;

  __shared__ __align__(16) bf16_t Es[2][32][40];   // [s][r]
  __shared__ __align__(16) bf16_t Fs[2][32][40];
  __shared__ __align__(16) bf16_t Ks[2][32][72];   // [s][d]
  __shared__ __align__(16) bf16_t Vs[2][32][72];

  const int tid  = threadIdx.x;
  const int lane = tid & 31;
  const int wid  = tid >> 5;
  const int hl   = lane >> 4, lr = lane & 15;
  const int mi   = wid & 1;            // r tile
  const int ni   = wid >> 1;           // d tile
  const int rI   = mi * 16 + lr;
  const int dI   = ni * 16 + lr;

  // 4 uniform async transfers per thread per tile
  auto stage = [&](int s0, int p) {
    int ss = tid >> 3;
    int c4 = (tid & 7) * 4;
    int c8 = (tid & 7) * 8;
    size_t rowEF = (size_t)(b * CS + s0 + ss) * CHR + h * CR;
    size_t rowKV = (size_t)(b * CS + s0 + ss) * CHK + h * CDK;
    cp8 (&Es[p][ss][c4], eh + rowEF + c4);
    cp8 (&Fs[p][ss][c4], fh + rowEF + c4);
    cp16(&Ks[p][ss][c8], kh + rowKV + c8);
    cp16(&Vs[p][ss][c8], vh + rowKV + c8);
  };

  v8f ak = {}, av = {};
  stage(0, 0);

  for (int s0 = 0; s0 < CS; s0 += 32) {
    const int p = (s0 >> 5) & 1;
    if (s0 + 32 < CS) { stage(s0 + 32, 1 - p); WAIT_ASYNC(4); }
    else              { WAIT_ASYNC(0); }
    __syncthreads();

    const int ka = hl * 16;
    const int kb = hl * 16;
    v16bf ae, af, bk, bv2;
#pragma unroll
    for (int j = 0; j < 4; ++j) {
      ae[2 * j]         = Es[p][ka + 2 * j][rI];
      ae[2 * j + 1]     = Es[p][ka + 2 * j + 1][rI];
      ae[8 + 2 * j]     = Es[p][ka + 8 + 2 * j][rI];
      ae[8 + 2 * j + 1] = Es[p][ka + 8 + 2 * j + 1][rI];
      af[2 * j]         = Fs[p][ka + 2 * j][rI];
      af[2 * j + 1]     = Fs[p][ka + 2 * j + 1][rI];
      af[8 + 2 * j]     = Fs[p][ka + 8 + 2 * j][rI];
      af[8 + 2 * j + 1] = Fs[p][ka + 8 + 2 * j + 1][rI];
    }
#pragma unroll
    for (int j = 0; j < 8; ++j) {
      bk[2 * j]      = Ks[p][kb + 2 * j][dI];
      bk[2 * j + 1]  = Ks[p][kb + 2 * j + 1][dI];
      bv2[2 * j]     = Vs[p][kb + 2 * j][dI];
      bv2[2 * j + 1] = Vs[p][kb + 2 * j + 1][dI];
    }
    ak = __builtin_amdgcn_wmma_f32_16x16x32_bf16(false, ae, false, bk,
                                                 (short)0, ak, false, false);
    av = __builtin_amdgcn_wmma_f32_16x16x32_bf16(false, af, false, bv2,
                                                 (short)0, av, false, false);
    __syncthreads();
  }

#pragma unroll
  for (int v = 0; v < 8; ++v) {
    int r = mi * 16 + v + hl * 8;
    size_t o = ((size_t)(b * CH + h) * CR + r) * CDK + dI;
    kc[o] = (bf16_t)ak[v];
    vc[o] = (bf16_t)av[v];
  }
}

// ---------------------------------------------------------------------------
// Attention over compressed axis. Workgroup = (b, h, 128 query rows).
// kc/vc tiles async-staged once in natural [r][d] order.
// ---------------------------------------------------------------------------
__global__ __launch_bounds__(256)
void linformer_attn_wmma(const bf16_t* __restrict__ qh,
                         const bf16_t* __restrict__ kc,
                         const bf16_t* __restrict__ vc,
                         float* __restrict__ scores_out,
                         bf16_t* __restrict__ oh) {
  const int b = blockIdx.z, h = blockIdx.y;
  const int q_blk = blockIdx.x * 128;

  __shared__ __align__(16) bf16_t KCs[32][72];     // [r][d] : B for scores (k=d)
  __shared__ __align__(16) bf16_t VCs[32][72];     // [r][d] : B for output (k=r)
  __shared__ float  Sc[8][16][32];                 // raw scores per wave
  __shared__ __align__(16) bf16_t Aw[8][16][40];   // softmaxed attn (A matrix)

  const int tid  = threadIdx.x;
  const int lane = tid & 31;
  const int wid  = tid >> 5;
  const int hl   = lane >> 4, lr = lane & 15;
  const int m0   = q_blk + wid * 16;

  {
    int rr = tid >> 3, d8 = (tid & 7) * 8;
    size_t o = ((size_t)(b * CH + h) * CR + rr) * CDK + d8;
    cp16(&KCs[rr][d8], kc + o);
    cp16(&VCs[rr][d8], vc + o);
  }
  WAIT_ASYNC(0);
  __syncthreads();

  // ---- scores = qh (16x64) * kc^T (64x32) ----
  v8f accs[2] = {{}, {}};
  const bf16_t* qrow = qh + (size_t)(b * CS + m0 + lr) * CHK + h * CDK;
  const int ka = hl * 16;
  const int kb = hl * 16;
#pragma unroll
  for (int kt = 0; kt < 2; ++kt) {
    v16bf a;
#pragma unroll
    for (int j = 0; j < 4; ++j) {
      a[2 * j]         = qrow[kt * 32 + ka + 2 * j];
      a[2 * j + 1]     = qrow[kt * 32 + ka + 2 * j + 1];
      a[8 + 2 * j]     = qrow[kt * 32 + ka + 8 + 2 * j];
      a[8 + 2 * j + 1] = qrow[kt * 32 + ka + 8 + 2 * j + 1];
    }
#pragma unroll
    for (int nt = 0; nt < 2; ++nt) {
      v16bf bb;
#pragma unroll
      for (int j = 0; j < 8; ++j) {
        bb[2 * j]     = KCs[nt * 16 + lr][kt * 32 + kb + 2 * j];
        bb[2 * j + 1] = KCs[nt * 16 + lr][kt * 32 + kb + 2 * j + 1];
      }
      accs[nt] = __builtin_amdgcn_wmma_f32_16x16x32_bf16(
          false, a, false, bb, (short)0, accs[nt], false, false);
    }
  }

  // ---- scale, emit raw scores, park in LDS for softmax ----
  const float inv_sqrt_dk = 0.125f;   // 1/sqrt(64)
#pragma unroll
  for (int nt = 0; nt < 2; ++nt) {
    int r = nt * 16 + lr;
#pragma unroll
    for (int v = 0; v < 8; ++v) {
      int qi = m0 + v + hl * 8;
      float sv = accs[nt][v] * inv_sqrt_dk;
      scores_out[((size_t)(b * CH + h) * CS + qi) * CR + r] = sv;
      Sc[wid][v + hl * 8][r] = sv;
    }
  }
  __syncthreads();

  // ---- row softmax over R=32 (lanes 0-15 each own one row) ----
  if (lane < 16) {
    float m = -3.4e38f;
#pragma unroll
    for (int c = 0; c < 32; ++c) m = fmaxf(m, Sc[wid][lr][c]);
    float ssum = 0.f;
#pragma unroll
    for (int c = 0; c < 32; ++c) {
      float e = __expf(Sc[wid][lr][c] - m);
      ssum += e;
      Aw[wid][lr][c] = (bf16_t)e;
    }
    float inv = 1.f / ssum;
#pragma unroll
    for (int c = 0; c < 32; ++c)
      Aw[wid][lr][c] = (bf16_t)((float)Aw[wid][lr][c] * inv);
  }
  __syncthreads();

  // ---- out = attn (16x32) * vc (32x64) ----
  v16bf a2;
#pragma unroll
  for (int j = 0; j < 4; ++j) {
    a2[2 * j]         = Aw[wid][lr][ka + 2 * j];
    a2[2 * j + 1]     = Aw[wid][lr][ka + 2 * j + 1];
    a2[8 + 2 * j]     = Aw[wid][lr][ka + 8 + 2 * j];
    a2[8 + 2 * j + 1] = Aw[wid][lr][ka + 8 + 2 * j + 1];
  }
#pragma unroll
  for (int nt = 0; nt < 4; ++nt) {
    const int d = nt * 16 + lr;
    v16bf b2;
#pragma unroll
    for (int j = 0; j < 8; ++j) {
      b2[2 * j]     = VCs[kb + 2 * j][d];       // B[k=r][n=d] = vc[r][d]
      b2[2 * j + 1] = VCs[kb + 2 * j + 1][d];
    }
    v8f o = {};
    o = __builtin_amdgcn_wmma_f32_16x16x32_bf16(false, a2, false, b2,
                                                (short)0, o, false, false);
#pragma unroll
    for (int v = 0; v < 8; ++v) {
      int qi = m0 + v + hl * 8;
      oh[(size_t)(b * CS + qi) * CHK + h * CDK + d] = (bf16_t)o[v];
    }
  }
}

// ---------------------------------------------------------------------------
extern "C" void kernel_launch(void* const* d_in, const int* in_sizes, int n_in,
                              void* d_out, int out_size, void* d_ws, size_t ws_size,
                              hipStream_t stream) {
  (void)in_sizes; (void)n_in; (void)out_size; (void)ws_size;

  const float* q  = (const float*)d_in[0];
  const float* k  = (const float*)d_in[1];
  const float* v  = (const float*)d_in[2];
  const float* e  = (const float*)d_in[3];
  const float* f  = (const float*)d_in[4];
  const float* Wq = (const float*)d_in[5];
  const float* bq = (const float*)d_in[6];
  const float* Wk = (const float*)d_in[7];
  const float* bk = (const float*)d_in[8];
  const float* Wv = (const float*)d_in[9];
  const float* bv = (const float*)d_in[10];
  const float* We = (const float*)d_in[11];
  const float* be = (const float*)d_in[12];
  const float* Wf = (const float*)d_in[13];
  const float* bfv = (const float*)d_in[14];
  const float* Wo = (const float*)d_in[15];
  const float* bo = (const float*)d_in[16];

  float* out_main   = (float*)d_out;                          // (B,S,D)
  float* out_scores = (float*)d_out + (size_t)CB * CS * CD;   // (B,H,S,R)

  char* ws = (char*)d_ws;
  size_t off = 0;
  auto take = [&](size_t elems) -> bf16_t* {
    bf16_t* p = (bf16_t*)(ws + off);
    off += (elems * sizeof(bf16_t) + 255) & ~(size_t)255;
    return p;
  };
  bf16_t* wq = take((size_t)CD * CHK);
  bf16_t* wk = take((size_t)CD * CHK);
  bf16_t* wv = take((size_t)CD * CHK);
  bf16_t* we = take((size_t)CD * CHR);
  bf16_t* wf = take((size_t)CD * CHR);
  bf16_t* wo = take((size_t)CHK * CD);
  bf16_t* qh = take((size_t)CM * CHK);
  bf16_t* kh = take((size_t)CM * CHK);
  bf16_t* vh = take((size_t)CM * CHK);
  bf16_t* eh = take((size_t)CM * CHR);
  bf16_t* fh = take((size_t)CM * CHR);
  bf16_t* oh = take((size_t)CM * CHK);
  bf16_t* kc = take((size_t)CB * CH * CR * CDK);
  bf16_t* vc = take((size_t)CB * CH * CR * CDK);

  // 1. weights f32 -> bf16
  auto cvt = [&](const float* src, bf16_t* dst, int n) {
    f32_to_bf16_kernel<<<(n + 255) / 256, 256, 0, stream>>>(src, dst, n);
  };
  cvt(Wq, wq, CD * CHK);
  cvt(Wk, wk, CD * CHK);
  cvt(Wv, wv, CD * CHK);
  cvt(We, we, CD * CHR);
  cvt(Wf, wf, CD * CHR);
  cvt(Wo, wo, CHK * CD);

  // 2. input projections (bf16 WMMA, f32 accumulate)
  dim3 gHK(CHK / 64, CM / 64);
  dim3 gHR(CHR / 64, CM / 64);
  gemm_bias_wmma<false, true><<<gHK, 256, 0, stream>>>(q, wq, bq, qh, CM, CHK, CD);
  gemm_bias_wmma<false, true><<<gHK, 256, 0, stream>>>(k, wk, bk, kh, CM, CHK, CD);
  gemm_bias_wmma<false, true><<<gHK, 256, 0, stream>>>(v, wv, bv, vh, CM, CHK, CD);
  gemm_bias_wmma<false, true><<<gHR, 256, 0, stream>>>(e, we, be, eh, CM, CHR, CD);
  gemm_bias_wmma<false, true><<<gHR, 256, 0, stream>>>(f, wf, bfv, fh, CM, CHR, CD);

  // 3. Linformer compression (K = S = 4096)
  compress_kv_wmma<<<CB * CH, 256, 0, stream>>>(eh, fh, kh, vh, kc, vc);

  // 4. attention over compressed axis + raw scores output
  dim3 gAtt(CS / 128, CH, CB);
  linformer_attn_wmma<<<gAtt, 256, 0, stream>>>(qh, kc, vc, out_scores, oh);

  // 5. output projection -> d_out (f32)
  gemm_bias_wmma<true, false><<<gHK, 256, 0, stream>>>(oh, wo, bo, out_main,
                                                       CM, CD, CHK);
}
